// SwinTransformerBlock_47459388621551
// MI455X (gfx1250) — matrix operand order repaired and checked by reading, not compiled, not run on hardware
//
#include <hip/hip_runtime.h>

// ---------------------------------------------------------------------------
// Types / helpers
// ---------------------------------------------------------------------------
typedef unsigned short u16;
typedef __attribute__((ext_vector_type(16))) __bf16 v16bf;
typedef __attribute__((ext_vector_type(8)))  float  v8f;

union Frag16 {
    uint4 u4[2];
    u16   us[16];
    v16bf v;
};

__device__ inline u16 f2bf(float f) {
    union { float f; unsigned int u; } c; c.f = f;
    unsigned int u = c.u;
    if ((u & 0x7fffffffu) > 0x7f800000u) return (u16)((u >> 16) | 0x0040u); // qNaN
    u += 0x7fffu + ((u >> 16) & 1u);   // round-to-nearest-even
    return (u16)(u >> 16);
}

__device__ inline v16bf load_frag_2x16B(const u16* p0, const u16* p1) {
    Frag16 f;
    f.u4[0] = *(const uint4*)p0;
    f.u4[1] = *(const uint4*)p1;
    return f.v;
}

__device__ inline v16bf zero_frag16() {
    Frag16 f;
    f.u4[0] = make_uint4(0u, 0u, 0u, 0u);
    f.u4[1] = make_uint4(0u, 0u, 0u, 0u);
    return f.v;
}

__device__ inline v8f zero8() {
    v8f z = {0.f, 0.f, 0.f, 0.f, 0.f, 0.f, 0.f, 0.f};
    return z;
}

__device__ inline v8f wmma_bf16(v16bf a, v16bf b, v8f c) {
    // D = A(16x32 bf16) * B(32x16 bf16) + C(16x16 f32)
    return __builtin_amdgcn_wmma_f32_16x16x32_bf16(
        false, a, false, b, (short)0, c, false, false);
}

// CDNA5 async global->LDS copy (ASYNCcnt-tracked, no VGPR staging).
// lds_off: wave-relative LDS byte offset (low 32 bits of generic pointer).
__device__ inline void async_load_lds_b128(unsigned lds_off, const void* gptr) {
    asm volatile("global_load_async_to_lds_b128 %0, %1, off"
                 :: "v"(lds_off), "v"(gptr) : "memory");
}
__device__ inline unsigned lds_off32(const void* p) {
    return (unsigned)(uintptr_t)p;
}

// ---------------------------------------------------------------------------
// Problem constants
// ---------------------------------------------------------------------------
#define TOK   100352   // 32 * 56 * 56
#define CDIM  512
#define NHEAD 16
#define HDIM  32
#define NWIN  2048     // 32 * 8 * 8
#define ATT_SCALE 0.17677669529663687f  // 1/sqrt(32)

// workspace layout (bytes)
#define SZ_QKV   ((size_t)TOK * 1536 * 2)
#define SZ_ATTN  ((size_t)TOK * 512 * 2)
#define OFF_QKV  ((size_t)0)
#define OFF_ATTN (SZ_QKV)
#define OFF_MID  ((size_t)0)                       // reuses qkv+attn region
#define OFF_H    (SZ_QKV + SZ_ATTN)                // bf16 T*512 (ln1 then ln2 out)
#define OFF_X2   (OFF_H  + (size_t)TOK * 512 * 2)  // f32  T*512
#define OFF_WQ   (OFF_X2 + (size_t)TOK * 512 * 4)
#define OFF_WP   (OFF_WQ + (size_t)512 * 1536 * 2)
#define OFF_W1   (OFF_WP + (size_t)512 * 512  * 2)
#define OFF_W2   (OFF_W1 + (size_t)512 * 2048 * 2)

// ---------------------------------------------------------------------------
// Kernel: transpose + convert weight fp32 (K,N) -> bf16 (N,K)
// ---------------------------------------------------------------------------
__global__ __launch_bounds__(256) void wt_transpose_bf16(
        const float* __restrict__ W, u16* __restrict__ WT, int K, int N) {
    int idx = blockIdx.x * 256 + threadIdx.x;   // index into WT (N*K)
    if (idx >= N * K) return;
    int n = idx / K;
    int k = idx - n * K;
    WT[idx] = f2bf(W[(size_t)k * N + n]);
}

// ---------------------------------------------------------------------------
// Kernel: LayerNorm over C=512, fp32 in -> bf16 out. One wave per token.
// ---------------------------------------------------------------------------
__global__ __launch_bounds__(256) void layernorm_bf16(
        const float* __restrict__ x, const float* __restrict__ g,
        const float* __restrict__ b, u16* __restrict__ out, int T) {
    int wave = threadIdx.x >> 5;
    int lane = threadIdx.x & 31;
    int t = blockIdx.x * 8 + wave;
    if (t >= T) return;

    const float4* xr = (const float4*)(x + (size_t)t * CDIM);
    float4 v[4];
    float s = 0.f;
#pragma unroll
    for (int j = 0; j < 4; ++j) {
        v[j] = xr[j * 32 + lane];
        s += v[j].x + v[j].y + v[j].z + v[j].w;
    }
#pragma unroll
    for (int m = 16; m >= 1; m >>= 1) s += __shfl_xor(s, m, 32);
    float mu = s * (1.0f / CDIM);

    float q = 0.f;
#pragma unroll
    for (int j = 0; j < 4; ++j) {
        float dx = v[j].x - mu, dy = v[j].y - mu, dz = v[j].z - mu, dw = v[j].w - mu;
        q += dx * dx + dy * dy + dz * dz + dw * dw;
    }
#pragma unroll
    for (int m = 16; m >= 1; m >>= 1) q += __shfl_xor(q, m, 32);
    float rstd = rsqrtf(q * (1.0f / CDIM) + 1e-6f);

#pragma unroll
    for (int j = 0; j < 4; ++j) {
        int c0 = (j * 32 + lane) * 4;
        ushort4 o;
        o.x = f2bf((v[j].x - mu) * rstd * g[c0 + 0] + b[c0 + 0]);
        o.y = f2bf((v[j].y - mu) * rstd * g[c0 + 1] + b[c0 + 1]);
        o.z = f2bf((v[j].z - mu) * rstd * g[c0 + 2] + b[c0 + 2]);
        o.w = f2bf((v[j].w - mu) * rstd * g[c0 + 3] + b[c0 + 3]);
        *(ushort4*)(out + (size_t)t * CDIM + c0) = o;
    }
}

// ---------------------------------------------------------------------------
// Kernel: bf16 WMMA GEMM  C[M,N] = A[M,K] * B[K,N] + bias, fused epilogue.
//   A : bf16 row-major (M,K),  BT : bf16 row-major (N,K)  (pre-transposed B)
//   block tile 128x128, 8 waves (2x4), wave tile 64x32, K step 32.
//   Double-buffered LDS staging via GLOBAL_LOAD_ASYNC_TO_LDS_B128 (ASYNCcnt).
// ---------------------------------------------------------------------------
#define LDSK 40   // padded K-stride in LDS (elements); 80B rows keep 16B align

__global__ __launch_bounds__(256) void gemm_bf16_wmma(
        const u16* __restrict__ A, const u16* __restrict__ BT,
        const float* __restrict__ bias, const float* __restrict__ resid,
        float* __restrict__ outF, u16* __restrict__ outB,
        int N, int K, int gelu) {
    __shared__ __align__(16) u16 As[2][128 * LDSK];
    __shared__ __align__(16) u16 Bs[2][128 * LDSK];

    const int tid    = threadIdx.x;
    const int lane   = tid & 31;
    const int lanelo = lane & 15;
    const int lanehi = lane >> 4;
    const int wave   = tid >> 5;
    const int wm     = wave >> 2;   // 0..1 : 64-row slab
    const int wn     = wave & 3;    // 0..3 : 32-col slab
    const int bm     = blockIdx.y;  // M/128 blocks
    const int bn     = blockIdx.x;  // N/128 blocks

    // per-thread staging chunks: chunk = it*256+tid -> row r, 8-elem kk slot
    const int r0  = tid >> 2;                 // it = 0
    const int kk0 = (tid & 3) * 8;
    const int r1  = (256 + tid) >> 2;         // it = 1
    const int kk1 = kk0;                      // (chunk&3) identical

    // issue one 128x32 A tile + 128x32 B^T tile into LDS buffer `buf`
    auto issue_tile = [&](int k0, int buf) {
        async_load_lds_b128(lds_off32(&As[buf][r0 * LDSK + kk0]),
                            A  + (size_t)(bm * 128 + r0) * K + k0 + kk0);
        async_load_lds_b128(lds_off32(&Bs[buf][r0 * LDSK + kk0]),
                            BT + (size_t)(bn * 128 + r0) * K + k0 + kk0);
        async_load_lds_b128(lds_off32(&As[buf][r1 * LDSK + kk1]),
                            A  + (size_t)(bm * 128 + r1) * K + k0 + kk1);
        async_load_lds_b128(lds_off32(&Bs[buf][r1 * LDSK + kk1]),
                            BT + (size_t)(bn * 128 + r1) * K + k0 + kk1);
    };

    v8f acc[4][2];
#pragma unroll
    for (int mt = 0; mt < 4; ++mt)
#pragma unroll
        for (int nt = 0; nt < 2; ++nt) acc[mt][nt] = zero8();

    const int nk = K >> 5;
    issue_tile(0, 0);

    for (int ki = 0; ki < nk; ++ki) {
        const int buf = ki & 1;
        if (ki + 1 < nk) {
            issue_tile((ki + 1) << 5, buf ^ 1);
            // async loads retire in order: wait until only tile (ki+1)'s
            // 4 transfers remain -> tile ki fully in LDS.
            asm volatile("s_wait_asynccnt 0x4" ::: "memory");
        } else {
            asm volatile("s_wait_asynccnt 0x0" ::: "memory");
        }
        __syncthreads();

        // fragments per documented 16-bit layouts
        v16bf af[4];
#pragma unroll
        for (int mt = 0; mt < 4; ++mt) {
            const u16* base = &As[buf][(wm * 64 + mt * 16 + lanelo) * LDSK];
            af[mt] = load_frag_2x16B(base + lanehi * 8, base + 16 + lanehi * 8);
        }
        v16bf bf[2];
#pragma unroll
        for (int nt = 0; nt < 2; ++nt) {
            const u16* base = &Bs[buf][(wn * 32 + nt * 16 + lanelo) * LDSK + lanehi * 8];
            bf[nt] = load_frag_2x16B(base, base + 16);
        }
#pragma unroll
        for (int mt = 0; mt < 4; ++mt)
#pragma unroll
            for (int nt = 0; nt < 2; ++nt)
                acc[mt][nt] = wmma_bf16(af[mt], bf[nt], acc[mt][nt]);
        __syncthreads();   // everyone done reading `buf` before it is refilled
    }

    // epilogue: bias (+gelu) (+residual) -> f32 and/or bf16
#pragma unroll
    for (int mt = 0; mt < 4; ++mt) {
#pragma unroll
        for (int nt = 0; nt < 2; ++nt) {
            int col = bn * 128 + wn * 32 + nt * 16 + lanelo;
            float bb = bias[col];
#pragma unroll
            for (int i = 0; i < 8; ++i) {
                int row = bm * 128 + wm * 64 + mt * 16 + lanehi * 8 + i;
                float v = acc[mt][nt][i] + bb;
                if (gelu) v = 0.5f * v * (1.0f + erff(v * 0.7071067811865475f));
                size_t o = (size_t)row * N + col;
                if (resid) v += resid[o];
                if (outF) outF[o] = v;
                if (outB) outB[o] = f2bf(v);
            }
        }
    }
}

// ---------------------------------------------------------------------------
// Kernel: windowed MHA. One wave per (window, head). 49 tokens padded to 64.
//   qkv: bf16 (T, 1536) laid out [q|k|v] x [head*32+d]
// ---------------------------------------------------------------------------
#define PSTR 72   // padded LDS stride (elements); 144B rows keep 16B align

__global__ __launch_bounds__(128) void window_attention(
        const u16* __restrict__ qkv, u16* __restrict__ attnout) {
    __shared__ __align__(16) u16 smem[4 * 96 * PSTR];   // per wave: P(64) + VT(32)

    const int tid    = threadIdx.x;
    const int lane   = tid & 31;
    const int lanelo = lane & 15;
    const int lanehi = lane >> 4;
    const int wave   = tid >> 5;
    const int task   = blockIdx.x * 4 + wave;   // 0 .. 32767
    const int wi     = task >> 4;               // window id
    const int hd     = task & 15;               // head id

    u16* Pw  = smem + wave * 96 * PSTR;         // 64 x PSTR  (P, bf16)
    u16* VTw = Pw + 64 * PSTR;                  // 32 x PSTR  (V^T, bf16)

    const int wb = wi >> 6;                     // batch
    const int wh = (wi & 63) >> 3;              // window row
    const int ww = wi & 7;                      // window col
    const int tbase = wb * 3136 + wh * 7 * 56 + ww * 7;

    auto tok2row = [&](int j) -> int {          // j in [0,49)
        int r = j / 7;
        int c = j - r * 7;
        return tbase + r * 56 + c;
    };

    // ---- scores = Q * K^T -------------------------------------------------
    v16bf qf[4];
#pragma unroll
    for (int mt = 0; mt < 4; ++mt) {
        int m = mt * 16 + lanelo;
        if (m < 49) {
            const u16* p = qkv + (size_t)tok2row(m) * 1536 + hd * HDIM;
            qf[mt] = load_frag_2x16B(p + lanehi * 8, p + 16 + lanehi * 8);
        } else {
            qf[mt] = zero_frag16();
        }
    }
    v16bf kf[4];
#pragma unroll
    for (int nt = 0; nt < 4; ++nt) {
        int n = nt * 16 + lanelo;
        if (n < 49) {
            const u16* p = qkv + (size_t)tok2row(n) * 1536 + CDIM + hd * HDIM
                         + lanehi * 8;
            kf[nt] = load_frag_2x16B(p, p + 16);
        } else {
            kf[nt] = zero_frag16();
        }
    }
    v8f s[4][4];
#pragma unroll
    for (int mt = 0; mt < 4; ++mt)
#pragma unroll
        for (int nt = 0; nt < 4; ++nt) {
            s[mt][nt] = zero8();
            s[mt][nt] = wmma_bf16(qf[mt], kf[nt], s[mt][nt]);
        }

    // ---- softmax rows (row lives on one 16-lane half) ---------------------
#pragma unroll
    for (int mt = 0; mt < 4; ++mt) {
#pragma unroll
        for (int i = 0; i < 8; ++i) {
            float vv[4];
            float mx = -3.0e38f;
#pragma unroll
            for (int nt = 0; nt < 4; ++nt) {
                int col = nt * 16 + lanelo;
                float v = s[mt][nt][i] * ATT_SCALE;
                if (col >= 49) v = -3.0e38f;
                vv[nt] = v;
                mx = fmaxf(mx, v);
            }
#pragma unroll
            for (int m = 8; m >= 1; m >>= 1) mx = fmaxf(mx, __shfl_xor(mx, m, 32));
            float sum = 0.f;
#pragma unroll
            for (int nt = 0; nt < 4; ++nt) {
                vv[nt] = expf(vv[nt] - mx);
                sum += vv[nt];
            }
#pragma unroll
            for (int m = 8; m >= 1; m >>= 1) sum += __shfl_xor(sum, m, 32);
            float inv = 1.0f / sum;
            int prow = mt * 16 + lanehi * 8 + i;
#pragma unroll
            for (int nt = 0; nt < 4; ++nt)
                Pw[prow * PSTR + nt * 16 + lanelo] = f2bf(vv[nt] * inv);
        }
    }

    // ---- stage V^T in LDS (d-major so B fragments read contiguous) --------
    for (int idx = lane; idx < 64 * HDIM; idx += 32) {
        int d   = idx >> 6;       // 0..31
        int tok = idx & 63;       // 0..63
        u16 v = 0;
        if (tok < 49)
            v = qkv[(size_t)tok2row(tok) * 1536 + 2 * CDIM + hd * HDIM + d];
        VTw[d * PSTR + tok] = v;
    }
    __syncthreads();

    // ---- out = P * V ------------------------------------------------------
    v8f o[4][2];
#pragma unroll
    for (int mt = 0; mt < 4; ++mt)
#pragma unroll
        for (int nt = 0; nt < 2; ++nt) o[mt][nt] = zero8();

#pragma unroll
    for (int kt = 0; kt < 2; ++kt) {
        v16bf pa[4];
#pragma unroll
        for (int mt = 0; mt < 4; ++mt) {
            const u16* base = &Pw[(mt * 16 + lanelo) * PSTR + kt * 32];
            pa[mt] = load_frag_2x16B(base + lanehi * 8, base + 16 + lanehi * 8);
        }
        v16bf vb[2];
#pragma unroll
        for (int nt = 0; nt < 2; ++nt) {
            const u16* base = &VTw[(nt * 16 + lanelo) * PSTR + kt * 32 + lanehi * 8];
            vb[nt] = load_frag_2x16B(base, base + 16);
        }
#pragma unroll
        for (int mt = 0; mt < 4; ++mt)
#pragma unroll
            for (int nt = 0; nt < 2; ++nt)
                o[mt][nt] = wmma_bf16(pa[mt], vb[nt], o[mt][nt]);
    }

    // ---- scatter back to token-major (T, C) bf16 --------------------------
#pragma unroll
    for (int mt = 0; mt < 4; ++mt) {
#pragma unroll
        for (int nt = 0; nt < 2; ++nt) {
#pragma unroll
            for (int i = 0; i < 8; ++i) {
                int m = mt * 16 + lanehi * 8 + i;
                if (m < 49) {
                    size_t t = (size_t)tok2row(m);
                    attnout[t * CDIM + hd * HDIM + nt * 16 + lanelo] =
                        f2bf(o[mt][nt][i]);
                }
            }
        }
    }
}

// ---------------------------------------------------------------------------
// Host-side launch sequence
// ---------------------------------------------------------------------------
extern "C" void kernel_launch(void* const* d_in, const int* in_sizes, int n_in,
                              void* d_out, int out_size, void* d_ws, size_t ws_size,
                              hipStream_t stream) {
    const float* x      = (const float*)d_in[0];
    const float* ln1_g  = (const float*)d_in[1];
    const float* ln1_b  = (const float*)d_in[2];
    const float* qkv_w  = (const float*)d_in[3];
    const float* qkv_b  = (const float*)d_in[4];
    const float* proj_w = (const float*)d_in[5];
    const float* proj_b = (const float*)d_in[6];
    const float* ln2_g  = (const float*)d_in[7];
    const float* ln2_b  = (const float*)d_in[8];
    const float* mlp_w1 = (const float*)d_in[9];
    const float* mlp_b1 = (const float*)d_in[10];
    const float* mlp_w2 = (const float*)d_in[11];
    const float* mlp_b2 = (const float*)d_in[12];
    float* out = (float*)d_out;

    char* ws = (char*)d_ws;
    u16*   qkvbuf = (u16*)  (ws + OFF_QKV);
    u16*   attnb  = (u16*)  (ws + OFF_ATTN);
    u16*   midb   = (u16*)  (ws + OFF_MID);
    u16*   hbf    = (u16*)  (ws + OFF_H);
    float* x2     = (float*)(ws + OFF_X2);
    u16*   wqT    = (u16*)  (ws + OFF_WQ);
    u16*   wpT    = (u16*)  (ws + OFF_WP);
    u16*   w1T    = (u16*)  (ws + OFF_W1);
    u16*   w2T    = (u16*)  (ws + OFF_W2);

    // 1) one-time weight transpose+convert to bf16 (N,K) layout
    wt_transpose_bf16<<<(512 * 1536 + 255) / 256, 256, 0, stream>>>(qkv_w,  wqT, 512,  1536);
    wt_transpose_bf16<<<(512 * 512  + 255) / 256, 256, 0, stream>>>(proj_w, wpT, 512,  512);
    wt_transpose_bf16<<<(512 * 2048 + 255) / 256, 256, 0, stream>>>(mlp_w1, w1T, 512,  2048);
    wt_transpose_bf16<<<(2048 * 512 + 255) / 256, 256, 0, stream>>>(mlp_w2, w2T, 2048, 512);

    // 2) LN1 -> bf16
    layernorm_bf16<<<TOK / 8, 256, 0, stream>>>(x, ln1_g, ln1_b, hbf, TOK);

    // 3) QKV GEMM: (T,512) x (512,1536) -> bf16 qkv
    gemm_bf16_wmma<<<dim3(1536 / 128, TOK / 128), 256, 0, stream>>>(
        hbf, wqT, qkv_b, nullptr, nullptr, qkvbuf, 1536, 512, 0);

    // 4) windowed attention (2048 windows x 16 heads, wave per task)
    window_attention<<<(NWIN * NHEAD) / 4, 128, 0, stream>>>(qkvbuf, attnb);

    // 5) proj GEMM + residual(x) -> x2 (f32)
    gemm_bf16_wmma<<<dim3(512 / 128, TOK / 128), 256, 0, stream>>>(
        attnb, wpT, proj_b, x, x2, nullptr, 512, 512, 0);

    // 6) LN2 -> bf16 (reuse hbf region)
    layernorm_bf16<<<TOK / 8, 256, 0, stream>>>(x2, ln2_g, ln2_b, hbf, TOK);

    // 7) MLP1 GEMM + exact GELU -> bf16 mid (reuses qkv+attn region)
    gemm_bf16_wmma<<<dim3(2048 / 128, TOK / 128), 256, 0, stream>>>(
        hbf, w1T, mlp_b1, nullptr, nullptr, midb, 2048, 512, 1);

    // 8) MLP2 GEMM + residual(x2) -> out (f32)
    gemm_bf16_wmma<<<dim3(512 / 128, TOK / 128), 256, 0, stream>>>(
        midb, w2T, mlp_b2, x2, out, nullptr, 512, 2048, 0);
}
